// MHA_29944511988286
// MI455X (gfx1250) — matrix operand order repaired and checked by reading, not compiled, not run on hardware
//
#include <hip/hip_runtime.h>
#include <hip/hip_bf16.h>

// ---------------------------------------------------------------------------
// Causal MHA (S=T=4096, H=16, D=64) + output projection (1024x1024), MI455X.
// bf16 V_WMMA_F32_16X16X32_BF16, fp32 accumulation.
//   Kernel 1: flash-attention-2, 8 waves/block, 128 q-rows/block, 64-key tiles
//             in LDS. K natural [n][d], V transposed [d][n] => every fragment
//             is a contiguous 16B lane read (ds_load_b128). B-fragments are
//             pre-staged so WMMAs issue back-to-back. Scale & ln2 folded into
//             Q; causal mask only on diagonal tiles; future tiles skipped.
//   Kernel 2: Y = X @ W^T via WMMA, one 16x16 tile per wave, K=1024 loop.
// ---------------------------------------------------------------------------

typedef __attribute__((ext_vector_type(16))) __bf16 v16bf;
typedef __attribute__((ext_vector_type(8)))  __bf16 v8bf;
typedef __attribute__((ext_vector_type(4)))  __bf16 v4bf;
typedef __attribute__((ext_vector_type(8)))  float  v8f;

#define SEQ   4096
#define NHEAD 16
#define HDIM  64
#define NEMB  1024   // NHEAD*HDIM
#define BM    128    // query rows per block = 8 waves x 16
#define BN    64     // key tile
#define LDP   72     // padded LDS row: 144 B, 16B-aligned, conflict-free b128

union V16U { v16bf v; v8bf h[2]; };
union V16W { v16bf v; unsigned w[8]; };
union V4W  { v4bf  v; unsigned w[2]; };

static __device__ __forceinline__ v16bf join16(v8bf lo, v8bf hi) {
    V16U u; u.h[0] = lo; u.h[1] = hi; return u.v;
}

// pack two fp32 -> packed bf16 pair (round-half-up) with one v_perm_b32
static __device__ __forceinline__ unsigned pk2bf(float fa, float fb) {
    union { float f; unsigned u; } a, b;
    a.f = fa; b.f = fb;
    return __builtin_amdgcn_perm(b.u + 0x8000u, a.u + 0x8000u, 0x07060302u);
}

static __device__ __forceinline__ __bf16 f2bfr(float f) {
    union { float f; unsigned u; } a; a.f = f;
    unsigned short h = (unsigned short)((a.u + 0x8000u) >> 16);
    __bf16 b;
    __builtin_memcpy(&b, &h, sizeof(b));
    return b;
}

static __device__ __forceinline__ v8f wmma_bf16(v16bf a, v16bf b, v8f c) {
    return __builtin_amdgcn_wmma_f32_16x16x32_bf16(false, a, false, b,
                                                   (short)0, c, false, false);
}

// ---------------------------------------------------------------------------
// Kernel 1: attention.  grid = (SEQ/BM, NHEAD), block = 256 threads.
// ---------------------------------------------------------------------------
__global__ __launch_bounds__(256)
void attn_fwd_kernel(const float* __restrict__ q, const float* __restrict__ k,
                     const float* __restrict__ v, __bf16* __restrict__ attn_out)
{
    __shared__ __align__(16) __bf16 sK [BN  ][LDP];   // K tile natural [n][d]
    __shared__ __align__(16) __bf16 sVt[HDIM][LDP];   // V tile transposed [d][n]
    __shared__ __align__(16) __bf16 sP [8][16][LDP];  // per-wave P scratch [m][n]

    const int h    = blockIdx.y;
    const int m0   = blockIdx.x * BM;
    const int tid  = threadIdx.x;
    const int wave = tid >> 5;
    const int lane = tid & 31;
    const int half = lane >> 4;
    const int l16  = lane & 15;
    const int mrow = m0 + wave * 16;          // this wave's first query row

    // scale (1/sqrt(64)) and ln2 folded into Q so softmax uses raw exp2
    const float qscale = 0.125f * 1.44269504088896340736f;
    const float bias2  = -10000.0f * 1.44269504088896340736f;

    // ---- Q fragments (A-layout: per lane 8 contiguous at kb, 8 at kb+16)
    v16bf qa[2];
    {
        const float* qrow = q + (size_t)(mrow + l16) * NEMB + h * HDIM;
        #pragma unroll
        for (int c = 0; c < 2; ++c) {
            const int kb = c * 32 + half * 8;
            float4 f0 = *(const float4*)(qrow + kb);
            float4 f1 = *(const float4*)(qrow + kb + 4);
            float4 f2 = *(const float4*)(qrow + kb + 16);
            float4 f3 = *(const float4*)(qrow + kb + 20);
            V16W u;
            u.w[0] = pk2bf(f0.x * qscale, f0.y * qscale);
            u.w[1] = pk2bf(f0.z * qscale, f0.w * qscale);
            u.w[2] = pk2bf(f1.x * qscale, f1.y * qscale);
            u.w[3] = pk2bf(f1.z * qscale, f1.w * qscale);
            u.w[4] = pk2bf(f2.x * qscale, f2.y * qscale);
            u.w[5] = pk2bf(f2.z * qscale, f2.w * qscale);
            u.w[6] = pk2bf(f3.x * qscale, f3.y * qscale);
            u.w[7] = pk2bf(f3.z * qscale, f3.w * qscale);
            qa[c] = u.v;
        }
    }

    v8f o[4];
    #pragma unroll
    for (int dd = 0; dd < 4; ++dd) o[dd] = (v8f)0.0f;
    float mprev[8], lsum[8];
    #pragma unroll
    for (int r = 0; r < 8; ++r) { mprev[r] = -1e30f; lsum[r] = 0.0f; }

    const int jend = m0 + BM;                 // causal: keys [0, m0+BM)

    for (int j0 = 0; j0 < jend; j0 += BN) {
        __syncthreads();                      // previous tile fully consumed
        // ---- cooperative K/V fill: float4 global loads, K->ds_store_b64,
        //      V transposed with 4 b16 stores per float4
        #pragma unroll
        for (int it = 0; it < (BN * HDIM / 4) / 256; ++it) {
            const int e  = tid + it * 256;
            const int n  = e >> 4;            // 16 float4 per 64-wide row
            const int d4 = (e & 15) * 4;
            const size_t g = (size_t)(j0 + n) * NEMB + h * HDIM + d4;
            const float4 kk4 = *(const float4*)(k + g);
            const float4 vv4 = *(const float4*)(v + g);
            V4W kp;
            kp.w[0] = pk2bf(kk4.x, kk4.y);
            kp.w[1] = pk2bf(kk4.z, kk4.w);
            *(v4bf*)&sK[n][d4] = kp.v;
            sVt[d4 + 0][n] = f2bfr(vv4.x);
            sVt[d4 + 1][n] = f2bfr(vv4.y);
            sVt[d4 + 2][n] = f2bfr(vv4.z);
            sVt[d4 + 3][n] = f2bfr(vv4.w);
        }
        __syncthreads();

        // per-wave: skip tiles entirely in the causal future of this wave
        if (j0 <= mrow + 15) {
            // ---- S = Q K^T : pre-stage all 8 B-frags, then 8 WMMAs
            v16bf bk[8];
            #pragma unroll
            for (int t = 0; t < 4; ++t) {
                #pragma unroll
                for (int c = 0; c < 2; ++c) {
                    const __bf16* p = &sK[t * 16 + l16][c * 32 + half * 8];
                    bk[t * 2 + c] = join16(*(const v8bf*)p, *(const v8bf*)(p + 16));
                }
            }
            v8f s[4];
            #pragma unroll
            for (int t = 0; t < 4; ++t) {
                s[t] = wmma_bf16(qa[0], bk[t * 2 + 0], (v8f)0.0f);
                s[t] = wmma_bf16(qa[1], bk[t * 2 + 1], s[t]);
            }

            // causal bias only on diagonal-straddling tiles
            if (j0 + BN - 1 > mrow) {
                #pragma unroll
                for (int t = 0; t < 4; ++t) {
                    #pragma unroll
                    for (int r = 0; r < 8; ++r) {
                        const int mg = mrow + r + half * 8;
                        const int ng = j0 + t * 16 + l16;
                        s[t][r] += (mg < ng) ? bias2 : 0.0f;
                    }
                }
            }

            // ---- online softmax per row (base-2 domain)
            #pragma unroll
            for (int r = 0; r < 8; ++r) {
                float mx = fmaxf(fmaxf(s[0][r], s[1][r]), fmaxf(s[2][r], s[3][r]));
                #pragma unroll
                for (int d = 8; d >= 1; d >>= 1) mx = fmaxf(mx, __shfl_xor(mx, d));
                const float mnew = fmaxf(mprev[r], mx);
                const float p0 = exp2f(s[0][r] - mnew);
                const float p1 = exp2f(s[1][r] - mnew);
                const float p2 = exp2f(s[2][r] - mnew);
                const float p3 = exp2f(s[3][r] - mnew);
                float rs = (p0 + p1) + (p2 + p3);
                #pragma unroll
                for (int d = 8; d >= 1; d >>= 1) rs += __shfl_xor(rs, d);
                const float corr = exp2f(mprev[r] - mnew);
                lsum[r]  = lsum[r] * corr + rs;
                mprev[r] = mnew;
                #pragma unroll
                for (int dd = 0; dd < 4; ++dd) o[dd][r] *= corr;
                const int M = r + half * 8;
                sP[wave][M][ 0 + l16] = f2bfr(p0);
                sP[wave][M][16 + l16] = f2bfr(p1);
                sP[wave][M][32 + l16] = f2bfr(p2);
                sP[wave][M][48 + l16] = f2bfr(p3);
            }
            // same-wave LDS ops are in-order (DScnt); no barrier needed

            // ---- O += P V : pre-stage P (A) and V (B) frags, then 8 WMMAs
            v16bf pa[2], bv[8];
            #pragma unroll
            for (int c = 0; c < 2; ++c) {
                const __bf16* pb = &sP[wave][l16][c * 32 + half * 16];
                pa[c] = join16(*(const v8bf*)pb, *(const v8bf*)(pb + 8));
                #pragma unroll
                for (int dd = 0; dd < 4; ++dd) {
                    const __bf16* vb = &sVt[dd * 16 + l16][c * 32 + half * 16];
                    bv[c * 4 + dd] = join16(*(const v8bf*)vb, *(const v8bf*)(vb + 8));
                }
            }
            #pragma unroll
            for (int c = 0; c < 2; ++c)
                #pragma unroll
                for (int dd = 0; dd < 4; ++dd)
                    o[dd] = wmma_bf16(pa[c], bv[c * 4 + dd], o[dd]);
        }
    }

    // ---- normalize + store bf16 activations [mg][h*64 + d]
    #pragma unroll
    for (int r = 0; r < 8; ++r) {
        const float inv = 1.0f / lsum[r];
        const int mg = mrow + r + half * 8;
        #pragma unroll
        for (int dd = 0; dd < 4; ++dd) {
            attn_out[(size_t)mg * NEMB + h * HDIM + dd * 16 + l16] =
                f2bfr(o[dd][r] * inv);
        }
    }
}

// ---------------------------------------------------------------------------
// Kernel 2: Y = X @ W^T.  X bf16 [4096,1024], W fp32 [1024,1024], Y fp32.
// One 16x16 output tile per wave; 8 waves per block.
// ---------------------------------------------------------------------------
__global__ __launch_bounds__(256)
void proj_kernel(const __bf16* __restrict__ X, const float* __restrict__ W,
                 float* __restrict__ Y)
{
    const int tid  = threadIdx.x;
    const int wave = tid >> 5;
    const int lane = tid & 31;
    const int half = lane >> 4;
    const int l16  = lane & 15;

    const int wid = blockIdx.x * 8 + wave;     // 16384 wave tiles total
    const int m0  = (wid >> 6) * 16;           // 64 n-tiles per m-row
    const int n0  = (wid & 63) * 16;

    const __bf16* xrow = X + (size_t)(m0 + l16) * NEMB;   // A row M = l16
    const float*  wrow = W + (size_t)(n0 + l16) * NEMB;   // B col N = l16

    v8f acc = (v8f)0.0f;
    #pragma unroll 4
    for (int kc = 0; kc < NEMB; kc += 32) {
        // A: 16 contiguous bf16 per lane (kb+0..15)
        const __bf16* xp = xrow + kc + half * 16;
        const v16bf a = join16(*(const v8bf*)xp, *(const v8bf*)(xp + 8));

        // B: 16 contiguous fp32 per lane, convert with v_perm packing
        const float* wp = wrow + kc + half * 16;
        __builtin_prefetch(wp + 64, 0, 1);     // global_prefetch toward L2/L0
        float4 w0 = *(const float4*)(wp);
        float4 w1 = *(const float4*)(wp + 4);
        float4 w2 = *(const float4*)(wp + 8);
        float4 w3 = *(const float4*)(wp + 12);
        V16W u;
        u.w[0] = pk2bf(w0.x, w0.y); u.w[1] = pk2bf(w0.z, w0.w);
        u.w[2] = pk2bf(w1.x, w1.y); u.w[3] = pk2bf(w1.z, w1.w);
        u.w[4] = pk2bf(w2.x, w2.y); u.w[5] = pk2bf(w2.z, w2.w);
        u.w[6] = pk2bf(w3.x, w3.y); u.w[7] = pk2bf(w3.z, w3.w);

        acc = wmma_bf16(a, u.v, acc);
    }

    #pragma unroll
    for (int r = 0; r < 8; ++r)
        Y[(size_t)(m0 + r + half * 8) * NEMB + n0 + l16] = acc[r];
}

// ---------------------------------------------------------------------------
extern "C" void kernel_launch(void* const* d_in, const int* in_sizes, int n_in,
                              void* d_out, int out_size, void* d_ws, size_t ws_size,
                              hipStream_t stream)
{
    (void)in_sizes; (void)n_in; (void)out_size; (void)ws_size;
    const float* q  = (const float*)d_in[0];
    const float* k  = (const float*)d_in[1];
    const float* v  = (const float*)d_in[2];
    const float* Wp = (const float*)d_in[3];
    float*  out  = (float*)d_out;
    __bf16* attn = (__bf16*)d_ws;              // 4096*1024*2 = 8 MB scratch

    dim3 g1(SEQ / BM, NHEAD);                  // 32 x 16 blocks
    attn_fwd_kernel<<<g1, 256, 0, stream>>>(q, k, v, attn);

    const int tiles = (SEQ / 16) * (NEMB / 16);   // 16384 wave tiles
    proj_kernel<<<tiles / 8, 256, 0, stream>>>(attn, Wp, out);
}
